// GBF2DEncoder_78469052498660
// MI455X (gfx1250) — matrix operand order repaired and checked by reading, not compiled, not run on hardware
//
#include <hip/hip_runtime.h>
#include <math.h>

// Problem constants (from reference setup_inputs)
#define B_ 8
#define N_ 256
#define K_ 128
#define H_ 32
#define D_ 768

typedef __attribute__((ext_vector_type(16))) _Float16 v16h;
typedef __attribute__((ext_vector_type(8)))  _Float16 v8h;
typedef __attribute__((ext_vector_type(8)))  float    v8f;

#define SQRT_2PI_F 2.5066256595f  // sqrt(2*3.14159), per reference

// Prepacked B-fragment workspace layout (halves), after the 1 MB dist_sum region:
//   W1: l1_w  -> [kc:4][nt:8 ][lane:32][h:16]  = 16384 halves
//   W2: l2_w  -> [kc:4][nt:2 ][lane:32][h:16]  =  4096 halves
//   W3: ep_w  -> [kc:4][nt:48][lane:32][h:16]  = 98304 halves
#define W1_OFF   0
#define W2_OFF   16384
#define W3_OFF   20480
#define WTOT     118784
#define DISTSUM_BYTES ((size_t)B_ * N_ * K_ * sizeof(float))

__device__ __forceinline__ float gelu_exact(float x) {
    return 0.5f * x * (1.0f + erff(x * 0.70710678118f));
}

// Build a 16x32 f16 WMMA A-fragment from a row-major [rows][128] LDS tile.
// lanes 0-15 => M=lane, halves0-7 K=k0+0..7, halves8-15 K=k0+16..23
// lanes16-31 => M=lane-16, halves0-7 K=k0+8..15, halves8-15 K=k0+24..31
__device__ __forceinline__ v16h load_a_frag(const _Float16 (*tile)[K_], int lane, int k0) {
    int m  = lane & 15;
    int kb = k0 + ((lane & 16) ? 8 : 0);
    v8h lo = *(const v8h*)&tile[m][kb];
    v8h hi = *(const v8h*)&tile[m][kb + 16];
    v16h a;
#pragma unroll
    for (int u = 0; u < 8; ++u) { a[u] = lo[u]; a[8 + u] = hi[u]; }
    return a;
}

// ---------------------------------------------------------------------------
// Kernel 0: pack l1_w / l2_w / ep_w (row = out-channel, col = k) into
// WMMA-B-fragment layout, f16, once. 464 blocks x 256 = 118784 threads.
// ---------------------------------------------------------------------------
__global__ __launch_bounds__(256) void pack_weights_kernel(
    const float* __restrict__ l1_w, const float* __restrict__ l2_w,
    const float* __restrict__ ep_w, _Float16* __restrict__ packW)
{
    int idx = blockIdx.x * 256 + threadIdx.x;
    if (idx >= WTOT) return;
    const float* src;
    int nt, kc, rel;
    if (idx < W2_OFF) {                 // l1_w: nt 0..7
        rel = idx - W1_OFF;
        nt = (rel >> 9) & 7;  kc = rel >> 12;  src = l1_w;
    } else if (idx < W3_OFF) {          // l2_w: nt 0..1
        rel = idx - W2_OFF;
        nt = (rel >> 9) & 1;  kc = rel >> 10;  src = l2_w;
    } else {                            // ep_w: nt 0..47
        rel = idx - W3_OFF;
        int g = rel >> 9;               // kc*48 + nt
        nt = g % 48;  kc = g / 48;      src = ep_w;
    }
    int h  = rel & 15;
    int ln = (rel >> 4) & 31;
    int k    = kc * 32 + (h & 8) * 2 + (ln >> 4) * 8 + (h & 7);
    int cout = nt * 16 + (ln & 15);
    packW[idx] = (_Float16)src[(size_t)cout * K_ + k];
}

// ---------------------------------------------------------------------------
// Kernel 1: per-pair gbf -> GEMM1(l1,gelu) -> GEMM2(l2) -> attn (B,H,N,N)
// One block = 64 pairs (fixed b,i ; j0..j0+63). 256 threads = 8 waves.
// B-fragments come prepacked straight from global (L2-resident).
// ---------------------------------------------------------------------------
__global__ __launch_bounds__(256) void pair_attn_kernel(
    const float* __restrict__ pos, const float* __restrict__ means,
    const float* __restrict__ stds, const float* __restrict__ mul_w,
    const float* __restrict__ bias_w, const float* __restrict__ l1_b,
    const float* __restrict__ l2_b, const int* __restrict__ x,
    const int* __restrict__ nte, const _Float16* __restrict__ packW,
    float* __restrict__ out_attn)
{
    __shared__ alignas(64) _Float16 sm_gbf[64][K_];      // 16 KB
    __shared__ alignas(64) _Float16 sm_h[64][K_];        // 16 KB
    __shared__ float sm_attn[H_][64];                    // 8 KB
    __shared__ float sm_dist[64], sm_mul[64], sm_bias[64], sm_maskmul[64];
    __shared__ float sm_mean[K_], sm_istd[K_], sm_coef[K_];

    const int tid  = threadIdx.x;
    const int lane = tid & 31;
    const int wid  = tid >> 5;

    const int bid = blockIdx.x;
    const int jg = bid & 3;                 // 4 groups of 64 j's
    const int i  = (bid >> 2) & (N_ - 1);
    const int b  = bid >> 10;
    const int j0 = jg * 64;

    if (tid < K_) {
        float s = fabsf(stds[tid]) + 1e-5f;
        sm_mean[tid] = means[tid];
        sm_istd[tid] = 1.0f / s;
        sm_coef[tid] = 1.0f / (SQRT_2PI_F * s);
    }
    if (tid < 64) {
        const int j = j0 + tid;
        float dx = pos[(b * N_ + j) * 3 + 0] - pos[(b * N_ + i) * 3 + 0];
        float dy = pos[(b * N_ + j) * 3 + 1] - pos[(b * N_ + i) * 3 + 1];
        float dz = pos[(b * N_ + j) * 3 + 2] - pos[(b * N_ + i) * 3 + 2];
        float sq = dx * dx + dy * dy + dz * dz;
        sm_dist[tid] = (sq < 1e-4f) ? 100.0f : sqrtf(sq);
        size_t base = ((size_t)(b * N_ + i) * N_ + j) * 2;
        int e0 = nte[base + 0], e1 = nte[base + 1];
        sm_mul[tid]  = 0.5f * (mul_w[e0]  + mul_w[e1]);
        sm_bias[tid] = 0.5f * (bias_w[e0] + bias_w[e1]);
        int vi = (x[b * N_ + i] != 0);
        int vj = (x[b * N_ + j] != 0);
        sm_maskmul[tid] = (vi && vj) ? 0.0f : 1.0f;  // attn kept only on masked pairs
    }
    __syncthreads();

    // gbf: 64 rows x 128 k ; each thread: one row, 32 k's
    {
        const int m  = tid >> 2;
        const int ks = (tid & 3) * 32;
        const float xv = sm_mul[m] * sm_dist[m] + sm_bias[m];
#pragma unroll 8
        for (int u = 0; u < 32; ++u) {
            int k = ks + u;
            float z = (xv - sm_mean[k]) * sm_istd[k];
            sm_gbf[m][k] = (_Float16)(sm_coef[k] * __expf(-0.5f * z * z));
        }
    }
    __syncthreads();

    // GEMM1: wave owns h-tile nt = wid; loops over 4 m-tiles of 16 pairs
    {
        const int nt = wid;
        const int col = nt * 16 + (lane & 15);
        const float bv = l1_b[col];
        const int mb = (lane & 16) ? 8 : 0;
#pragma unroll
        for (int mt = 0; mt < 4; ++mt) {
            v8f c = {};
#pragma unroll
            for (int kc = 0; kc < 4; ++kc) {
                v16h a  = load_a_frag(&sm_gbf[mt * 16], lane, kc * 32);
                v16h bb = *(const v16h*)&packW[W1_OFF + (((kc * 8 + nt) * 32 + lane) << 4)];
                c = __builtin_amdgcn_wmma_f32_16x16x32_f16(
                        false, a, false, bb, (short)0, c, false, false);
            }
#pragma unroll
            for (int r = 0; r < 8; ++r)
                sm_h[mt * 16 + r + mb][col] = (_Float16)gelu_exact(c[r] + bv);
        }
    }
    __syncthreads();

    // GEMM2: all 8 waves active: wave = (m-tile, h-tile) pair
    {
        const int mt  = wid >> 1;
        const int nt  = wid & 1;
        const int hcol = nt * 16 + (lane & 15);
        const float bv = l2_b[hcol];
        const int mb = (lane & 16) ? 8 : 0;
        v8f c = {};
#pragma unroll
        for (int kc = 0; kc < 4; ++kc) {
            v16h a  = load_a_frag(&sm_h[mt * 16], lane, kc * 32);
            v16h bb = *(const v16h*)&packW[W2_OFF + (((kc * 2 + nt) * 32 + lane) << 4)];
            c = __builtin_amdgcn_wmma_f32_16x16x32_f16(
                    false, a, false, bb, (short)0, c, false, false);
        }
#pragma unroll
        for (int r = 0; r < 8; ++r)
            sm_attn[hcol][mt * 16 + r + mb] = c[r] + bv;
    }
    __syncthreads();

    // Coalesced masked store: attn[b][h][i][j0+jj], 64-float contiguous rows
#pragma unroll
    for (int s = 0; s < 8; ++s) {
        int idx = tid + 256 * s;            // 0..2047
        int hh = idx >> 6;
        int jj = idx & 63;
        float v = sm_attn[hh][jj] * sm_maskmul[jj];
        out_attn[(((size_t)b * H_ + hh) * N_ + i) * N_ + (j0 + jj)] = v;
    }
}

// ---------------------------------------------------------------------------
// Kernel 2: dist_sum[b,i,k] = sum_j maskmul * gbf(b,i,j,k)  (one block per (b,i))
// ---------------------------------------------------------------------------
__global__ __launch_bounds__(256) void dist_sum_kernel(
    const float* __restrict__ pos, const float* __restrict__ means,
    const float* __restrict__ stds, const float* __restrict__ mul_w,
    const float* __restrict__ bias_w, const int* __restrict__ x,
    const int* __restrict__ nte, float* __restrict__ dist_sum)
{
    __shared__ float sj_xv[N_], sj_mask[N_];
    __shared__ float sm_mean[K_], sm_istd[K_], sm_coef[K_];

    const int tid = threadIdx.x;
    const int bid = blockIdx.x;
    const int i = bid & (N_ - 1);
    const int b = bid >> 8;

    if (tid < K_) {
        float s = fabsf(stds[tid]) + 1e-5f;
        sm_mean[tid] = means[tid];
        sm_istd[tid] = 1.0f / s;
        sm_coef[tid] = 1.0f / (SQRT_2PI_F * s);
    }
    {
        const int j = tid;
        float dx = pos[(b * N_ + j) * 3 + 0] - pos[(b * N_ + i) * 3 + 0];
        float dy = pos[(b * N_ + j) * 3 + 1] - pos[(b * N_ + i) * 3 + 1];
        float dz = pos[(b * N_ + j) * 3 + 2] - pos[(b * N_ + i) * 3 + 2];
        float sq = dx * dx + dy * dy + dz * dz;
        float dist = (sq < 1e-4f) ? 100.0f : sqrtf(sq);
        size_t base = ((size_t)(b * N_ + i) * N_ + j) * 2;
        int e0 = nte[base + 0], e1 = nte[base + 1];
        float mul  = 0.5f * (mul_w[e0]  + mul_w[e1]);
        float bias = 0.5f * (bias_w[e0] + bias_w[e1]);
        sj_xv[j] = mul * dist + bias;
        int vi = (x[b * N_ + i] != 0);
        int vj = (x[b * N_ + j] != 0);
        sj_mask[j] = (vi && vj) ? 0.0f : 1.0f;
    }
    __syncthreads();

    if (tid < K_) {
        const float mean = sm_mean[tid], istd = sm_istd[tid], coef = sm_coef[tid];
        float acc = 0.0f;
#pragma unroll 4
        for (int j = 0; j < N_; ++j) {
            float z = (sj_xv[j] - mean) * istd;
            acc += sj_mask[j] * coef * __expf(-0.5f * z * z);
        }
        dist_sum[(size_t)bid * K_ + tid] = acc;
    }
}

// ---------------------------------------------------------------------------
// Kernel 3: dist_emb = (dist_sum[2048x128] x ep_w^T[128x768] + ep_b) / 100 via WMMA
// One block = 16 rows x 128 cols. grid = 128 * 6. B-fragments prepacked in global.
// ---------------------------------------------------------------------------
__global__ __launch_bounds__(256) void ep_proj_kernel(
    const float* __restrict__ dist_sum, const _Float16* __restrict__ packW,
    const float* __restrict__ ep_b, float* __restrict__ out_dist)
{
    __shared__ alignas(64) _Float16 sm_a[16][K_];        // 4 KB

    const int tid  = threadIdx.x;
    const int lane = tid & 31;
    const int wid  = tid >> 5;

    const int bid = blockIdx.x;
    const int mt = bid & 127;        // 2048/16 row tiles
    const int ng = bid >> 7;         // 768/128 col groups
    const int m0 = mt * 16, n0 = ng * 128;

    for (int it = 0; it < 8; ++it) {
        int idx = tid + 256 * it;    // 0..2047
        int r = idx >> 7, k = idx & 127;
        sm_a[r][k] = (_Float16)dist_sum[(size_t)(m0 + r) * K_ + k];
    }
    __syncthreads();

    {
        const int nt48 = ng * 8 + wid;              // 0..47 global n-tile
        v8f c = {};
#pragma unroll
        for (int kc = 0; kc < 4; ++kc) {
            v16h a  = load_a_frag(sm_a, lane, kc * 32);
            v16h bb = *(const v16h*)&packW[W3_OFF + (((kc * 48 + nt48) * 32 + lane) << 4)];
            c = __builtin_amdgcn_wmma_f32_16x16x32_f16(
                    false, a, false, bb, (short)0, c, false, false);
        }
        const int col  = n0 + wid * 16 + (lane & 15);
        const float bv = ep_b[col];
        const int mb   = (lane & 16) ? 8 : 0;
#pragma unroll
        for (int r = 0; r < 8; ++r)
            out_dist[(size_t)(m0 + r + mb) * D_ + col] = (c[r] + bv) * 0.01f;
    }
}

// ---------------------------------------------------------------------------
extern "C" void kernel_launch(void* const* d_in, const int* in_sizes, int n_in,
                              void* d_out, int out_size, void* d_ws, size_t ws_size,
                              hipStream_t stream) {
    const float* pos    = (const float*)d_in[0];
    const float* means  = (const float*)d_in[1];
    const float* stds   = (const float*)d_in[2];
    const float* mul_w  = (const float*)d_in[3];
    const float* bias_w = (const float*)d_in[4];
    const float* l1_w   = (const float*)d_in[5];
    const float* l1_b   = (const float*)d_in[6];
    const float* l2_w   = (const float*)d_in[7];
    const float* l2_b   = (const float*)d_in[8];
    const float* ep_w   = (const float*)d_in[9];
    const float* ep_b   = (const float*)d_in[10];
    const int*   x      = (const int*)d_in[11];
    const int*   nte    = (const int*)d_in[12];

    float* out_dist = (float*)d_out;                               // (B,N,D)
    float* out_attn = out_dist + (size_t)B_ * N_ * D_;             // (B,H,N,N)
    float*     dist_sum = (float*)d_ws;                            // 1 MB
    _Float16*  packW    = (_Float16*)((char*)d_ws + DISTSUM_BYTES);

    hipLaunchKernelGGL(pack_weights_kernel, dim3((WTOT + 255) / 256), dim3(256), 0, stream,
                       l1_w, l2_w, ep_w, packW);
    hipLaunchKernelGGL(pair_attn_kernel, dim3(B_ * N_ * (N_ / 64)), dim3(256), 0, stream,
                       pos, means, stds, mul_w, bias_w, l1_b, l2_b,
                       x, nte, packW, out_attn);
    hipLaunchKernelGGL(dist_sum_kernel, dim3(B_ * N_), dim3(256), 0, stream,
                       pos, means, stds, mul_w, bias_w, x, nte, dist_sum);
    hipLaunchKernelGGL(ep_proj_kernel, dim3((2048 / 16) * (D_ / 128)), dim3(256), 0, stream,
                       dist_sum, packW, ep_b, out_dist);
}
